// SplitQKVAttention_16320875724974
// MI455X (gfx1250) — matrix operand order repaired and checked by reading, not compiled, not run on hardware
//
#include <hip/hip_runtime.h>
#include <hip/hip_bf16.h>

// MI455X (gfx1250) fused multi-head attention, f16 WMMA / f32 accumulate.
//   pass 0: fp32 -> f16 conversion of activations and weights (done once,
//           instead of once per GEMM column-block; whole working set fits L2)
//   pass 1: 3x (8192x1024x1024) f16 GEMMs -> Q/K/V in (B,H,T,HD)
//   pass 2: flash-style attention (no TxT in HBM) -> f16 ctx (T*B, D)
//   pass 3: (8192x1024x1024) f16 GEMM -> fp32 out
// GEMM: 128x64 block tile, 8 waves x 4 WMMAs/chunk, double-buffered LDS,
// staged with GLOBAL_LOAD_ASYNC_TO_LDS_B128 (ASYNCcnt) when available.
// Workspace: 104 MiB.

typedef __attribute__((ext_vector_type(16))) _Float16 v16h;
typedef __attribute__((ext_vector_type(8)))  _Float16 v8h;
typedef __attribute__((ext_vector_type(8)))  float    v8f;
typedef __attribute__((ext_vector_type(4)))  int      v4i;

static constexpr int T_  = 1024;
static constexpr int B_  = 8;
static constexpr int D_  = 1024;
static constexpr int H_  = 16;
static constexpr int HD_ = 64;
static constexpr int M_  = T_ * B_;   // 8192 rows of the (T*B, D) view

#define WMMA_F16(a, b, c) \
  __builtin_amdgcn_wmma_f32_16x16x32_f16(false, (a), false, (b), (short)0, (c), false, false)

#if __has_builtin(__builtin_amdgcn_global_load_async_to_lds_b128) && \
    __has_builtin(__builtin_amdgcn_s_wait_asynccnt)
#define USE_ASYNC 1
#else
#define USE_ASYNC 0
#endif

#define AS1 __attribute__((address_space(1)))
#define AS3 __attribute__((address_space(3)))

#if USE_ASYNC
__device__ inline void async_b128(const _Float16* g, _Float16* l) {
  __builtin_amdgcn_global_load_async_to_lds_b128((AS1 v4i*)g, (AS3 v4i*)l, 0, 0);
}
#endif

// ---- fragment loaders (wave32; A and B share one contiguous k order, C/D per ISA) ----
// A 16x32 f16: lane m=lane&15; element i -> k = i + 16*(lane>>4)
__device__ inline v16h load_a_frag(const _Float16* base, int stride, int lane) {
  const int m = lane & 15, g = lane >> 4;
  v8h lo = *(const v8h*)(base + m * stride + 16 * g);
  v8h hi = *(const v8h*)(base + m * stride + 16 * g + 8);
  return __builtin_shufflevector(lo, hi, 0, 1, 2, 3, 4, 5, 6, 7,
                                 8, 9, 10, 11, 12, 13, 14, 15);
}
// B 32x16 f16: lane n=lane&15; element i -> k = i + 16*(lane>>4)  (16 contiguous halves)
__device__ inline v16h load_b_frag(const _Float16* base, int stride, int lane) {
  const int n = lane & 15, g = lane >> 4;
  return *(const v16h*)(base + n * stride + 16 * g);
}
// C/D 16x16 f32: element r -> row = r + 8*(lane>>4), col = lane&15.
__device__ inline float rowmax16(float x) {
  x = fmaxf(x, __shfl_xor(x, 1, 32));
  x = fmaxf(x, __shfl_xor(x, 2, 32));
  x = fmaxf(x, __shfl_xor(x, 4, 32));
  x = fmaxf(x, __shfl_xor(x, 8, 32));
  return x;
}
__device__ inline float rowsum16(float x) {
  x += __shfl_xor(x, 1, 32);
  x += __shfl_xor(x, 2, 32);
  x += __shfl_xor(x, 4, 32);
  x += __shfl_xor(x, 8, 32);
  return x;
}

// ------------------- fp32 -> f16 streaming conversion -------------------
__global__ __launch_bounds__(256) void cvt_f32_f16(const float* __restrict__ src,
                                                   _Float16* __restrict__ dst) {
  const int i = blockIdx.x * blockDim.x + threadIdx.x;   // one v8 group per thread
  const float4* s = (const float4*)src + 2 * (size_t)i;
  float4 f0 = s[0], f1 = s[1];
  v8h h = {(_Float16)f0.x, (_Float16)f0.y, (_Float16)f0.z, (_Float16)f0.w,
           (_Float16)f1.x, (_Float16)f1.y, (_Float16)f1.z, (_Float16)f1.w};
  *(v8h*)(dst + 8 * (size_t)i) = h;
}

// ------------------- GEMM: out = A @ W^T + bias (all-f16 operands) -------------------
// A: (8192 x 1024) f16, W: (1024 x 1024) f16 row-major.
// Block: 256 threads (8 waves), 128x64 output tile, K chunked by 32,
// double-buffered LDS, software-pipelined staging (async when available).
// OUT_MODE 0: scatter f16 to (B,H,T,HD);  OUT_MODE 1: fp32 to (T*B, D).
template <int OUT_MODE>
__global__ __launch_bounds__(256) void gemm_wmma(const _Float16* __restrict__ A,
                                                 const _Float16* __restrict__ W,
                                                 const float* __restrict__ bias,
                                                 void* __restrict__ outv) {
  __shared__ _Float16 ldsA[2][128][32];   // 16 KiB
  __shared__ _Float16 ldsB[2][64][32];    //  8 KiB

  const int tid   = threadIdx.x;
  const int lane  = tid & 31;
  const int w     = tid >> 5;             // wave 0..7 -> 16-row strip
  const int mbase = blockIdx.x * 128;
  const int nbase = blockIdx.y * 64;
  const int arow  = tid >> 1;             // 0..127
  const int aseg  = (tid & 1) * 16;       // 16 halves (32 B) per thread
  const int brow  = tid >> 2;             // 0..63
  const int bseg  = (tid & 3) * 8;        // 8 halves (16 B) per thread

  v8f acc0 = {}, acc1 = {}, acc2 = {}, acc3 = {};

  auto stage = [&](int buf, int k0) {
    const _Float16* ga = A + (size_t)(mbase + arow) * D_ + k0 + aseg;
    const _Float16* gb = W + (size_t)(nbase + brow) * D_ + k0 + bseg;
#if USE_ASYNC
    async_b128(ga,     &ldsA[buf][arow][aseg]);
    async_b128(ga + 8, &ldsA[buf][arow][aseg + 8]);
    async_b128(gb,     &ldsB[buf][brow][bseg]);
#else
    *(v8h*)&ldsA[buf][arow][aseg]     = *(const v8h*)ga;
    *(v8h*)&ldsA[buf][arow][aseg + 8] = *(const v8h*)(ga + 8);
    *(v8h*)&ldsB[buf][brow][bseg]     = *(const v8h*)gb;
#endif
  };

  stage(0, 0);
  constexpr int NK = D_ / 32;   // 32 chunks
  for (int i = 0; i < NK; ++i) {
    const int buf = i & 1;
    if (i + 1 < NK) {
      stage(buf ^ 1, (i + 1) * 32);     // overlaps next chunk with this WMMA
#if USE_ASYNC
      __builtin_amdgcn_s_wait_asynccnt(3);  // chunk i done (in-order), i+1 in flight
#endif
    } else {
#if USE_ASYNC
      __builtin_amdgcn_s_wait_asynccnt(0);
#endif
    }
    __syncthreads();
    v16h a = load_a_frag(&ldsA[buf][16 * w][0], 32, lane);
    acc0 = WMMA_F16(a, load_b_frag(&ldsB[buf][0][0],  32, lane), acc0);
    acc1 = WMMA_F16(a, load_b_frag(&ldsB[buf][16][0], 32, lane), acc1);
    acc2 = WMMA_F16(a, load_b_frag(&ldsB[buf][32][0], 32, lane), acc2);
    acc3 = WMMA_F16(a, load_b_frag(&ldsB[buf][48][0], 32, lane), acc3);
    __syncthreads();
  }

  const int g = lane >> 4, ncol = lane & 15;
#pragma unroll
  for (int ct = 0; ct < 4; ++ct) {
    v8f acc = (ct == 0) ? acc0 : (ct == 1) ? acc1 : (ct == 2) ? acc2 : acc3;
    const int ng = nbase + 16 * ct + ncol;
    const float bv = bias[ng];
#pragma unroll
    for (int r = 0; r < 8; ++r) {
      const int mg = mbase + 16 * w + r + 8 * g;   // row in (T*B)
      const float val = acc[r] + bv;
      if constexpr (OUT_MODE == 0) {
        _Float16* outh = (_Float16*)outv;
        const int t = mg >> 3, b = mg & 7;         // m = t*B + b
        const int h = ng >> 6, hd = ng & 63;       // n = h*HD + hd
        outh[(((size_t)b * H_ + h) * T_ + t) * HD_ + hd] = (_Float16)val;
      } else {
        float* outf = (float*)outv;
        outf[(size_t)mg * D_ + ng] = val;
      }
    }
  }
}

// ------------------- flash-style attention -------------------
// grid: (T/64, B*H), block: 128 threads (4 waves). Each wave owns 16 q-rows,
// streams keys/values in 32-wide chunks with online softmax in f32.
__global__ __launch_bounds__(128) void attn_wmma(const _Float16* __restrict__ qh,
                                                 const _Float16* __restrict__ kh,
                                                 const _Float16* __restrict__ vh,
                                                 _Float16* __restrict__ ctx) {
  __shared__ _Float16 ldsV[64][32];       // [hd][key-in-chunk] (transposed V)
  __shared__ _Float16 pbuf[4][16][32];    // per-wave probability tile (A-layout source)

  const int tid  = threadIdx.x;
  const int lane = tid & 31;
  const int w    = tid >> 5;
  const int g    = lane >> 4, ncol = lane & 15;
  const int bh   = blockIdx.y;            // b*H + h
  const int b    = bh >> 4, h = bh & 15;
  const int qbase = blockIdx.x * 64 + w * 16;

  const _Float16* qb = qh + ((size_t)bh * T_ + qbase) * HD_;
  const _Float16* kt = kh + (size_t)bh * T_ * HD_;
  const _Float16* vt = vh + (size_t)bh * T_ * HD_;

  // Q fragments resident for the whole pass (HD=64 -> two K=32 A-fragments)
  v16h qa0 = load_a_frag(qb, HD_, lane);
  v16h qa1 = load_a_frag(qb + 32, HD_, lane);

  v8f o0 = {}, o1 = {}, o2 = {}, o3 = {};
  float mrow[8], lrow[8];
#pragma unroll
  for (int r = 0; r < 8; ++r) { mrow[r] = -1.0e30f; lrow[r] = 0.0f; }

  const int vrow = tid >> 2;        // key index within chunk 0..31
  const int vseg = (tid & 3) * 16;  // hd segment of 16

  for (int kc = 0; kc < T_; kc += 32) {
    __syncthreads();  // previous ldsV consumers done
    // stage V chunk transposed: ldsV[hd][kk]
    {
      const _Float16* src = vt + (size_t)(kc + vrow) * HD_ + vseg;
      v8h va = *(const v8h*)src;
      v8h vb = *(const v8h*)(src + 8);
#pragma unroll
      for (int j = 0; j < 8; ++j) {
        ldsV[vseg + j][vrow]     = va[j];
        ldsV[vseg + 8 + j][vrow] = vb[j];
      }
      if (kc + 32 < T_) {
        __builtin_prefetch(src + 32 * HD_, 0, 1);                    // next V chunk
        __builtin_prefetch(kt + (size_t)(kc + 32) * HD_ + lane, 0, 1);  // next K chunk
      }
    }
    // S tiles (16 q-rows x 32 keys) straight from global K
    v8f s0 = {}, s1 = {};
    {
      const _Float16* k0 = kt + (size_t)kc * HD_;
      s0 = WMMA_F16(qa0, load_b_frag(k0, HD_, lane), s0);
      s0 = WMMA_F16(qa1, load_b_frag(k0 + 32, HD_, lane), s0);
      const _Float16* k1 = kt + (size_t)(kc + 16) * HD_;
      s1 = WMMA_F16(qa0, load_b_frag(k1, HD_, lane), s1);
      s1 = WMMA_F16(qa1, load_b_frag(k1 + 32, HD_, lane), s1);
    }
    __syncthreads();  // ldsV ready; previous pbuf consumed

    // online softmax update (HD^-0.5 = 0.125)
#pragma unroll
    for (int r = 0; r < 8; ++r) {
      const float a0 = s0[r] * 0.125f;
      const float a1 = s1[r] * 0.125f;
      const float mx = rowmax16(fmaxf(a0, a1));
      const float mnew = fmaxf(mrow[r], mx);
      const float alpha = __expf(mrow[r] - mnew);
      const float p0 = __expf(a0 - mnew);
      const float p1 = __expf(a1 - mnew);
      lrow[r] = lrow[r] * alpha + rowsum16(p0 + p1);
      mrow[r] = mnew;
      o0[r] *= alpha; o1[r] *= alpha; o2[r] *= alpha; o3[r] *= alpha;
      pbuf[w][r + 8 * g][ncol]      = (_Float16)p0;
      pbuf[w][r + 8 * g][ncol + 16] = (_Float16)p1;
    }
    __syncthreads();  // pbuf visible

    // O += P @ V   (P: 16x32 A-frag, V: 32x16 B-frags from transposed LDS)
    v16h pa = load_a_frag(&pbuf[w][0][0], 32, lane);
    o0 = WMMA_F16(pa, load_b_frag(&ldsV[0][0],  32, lane), o0);
    o1 = WMMA_F16(pa, load_b_frag(&ldsV[16][0], 32, lane), o1);
    o2 = WMMA_F16(pa, load_b_frag(&ldsV[32][0], 32, lane), o2);
    o3 = WMMA_F16(pa, load_b_frag(&ldsV[48][0], 32, lane), o3);
  }

  // normalize and write ctx in (T*B, D) f16 layout for the output GEMM
#pragma unroll
  for (int r = 0; r < 8; ++r) {
    const int tq = qbase + r + 8 * g;
    const float inv = 1.0f / lrow[r];
    const size_t rowoff = ((size_t)tq * B_ + b) * D_ + (size_t)h * HD_;
    ctx[rowoff + 0  + ncol] = (_Float16)(o0[r] * inv);
    ctx[rowoff + 16 + ncol] = (_Float16)(o1[r] * inv);
    ctx[rowoff + 32 + ncol] = (_Float16)(o2[r] * inv);
    ctx[rowoff + 48 + ncol] = (_Float16)(o3[r] * inv);
  }
}

extern "C" void kernel_launch(void* const* d_in, const int* in_sizes, int n_in,
                              void* d_out, int out_size, void* d_ws, size_t ws_size,
                              hipStream_t stream) {
  (void)in_sizes; (void)n_in; (void)out_size; (void)ws_size;
  const float* query = (const float*)d_in[0];
  const float* key   = (const float*)d_in[1];
  const float* value = (const float*)d_in[2];
  const float* Wq = (const float*)d_in[3];
  const float* bq = (const float*)d_in[4];
  const float* Wk = (const float*)d_in[5];
  const float* bk = (const float*)d_in[6];
  const float* Wv = (const float*)d_in[7];
  const float* bv = (const float*)d_in[8];
  const float* Wo = (const float*)d_in[9];
  const float* bo = (const float*)d_in[10];

  const size_t MB = 1u << 20;
  const size_t wB = (size_t)D_ * D_ * sizeof(_Float16);     //  2 MiB
  char* ws = (char*)d_ws;                                   // needs 104 MiB
  _Float16* xq = (_Float16*)(ws);                 // dead after q-proj
  _Float16* xk = (_Float16*)(ws + 16 * MB);
  _Float16* xv = (_Float16*)(ws + 32 * MB);
  _Float16* qh = (_Float16*)(ws + 48 * MB);
  _Float16* kh = (_Float16*)(ws + 64 * MB);
  _Float16* vh = (_Float16*)(ws + 80 * MB);
  _Float16* wq16 = (_Float16*)(ws + 96 * MB);
  _Float16* wk16 = (_Float16*)(ws + 96 * MB + wB);
  _Float16* wv16 = (_Float16*)(ws + 96 * MB + 2 * wB);
  _Float16* wo16 = (_Float16*)(ws + 96 * MB + 3 * wB);
  _Float16* ctx  = (_Float16*)(ws);               // reuse xq region (16 MiB)

  const int actBlocks = (M_ * D_ / 8) / 256;   // 4096
  const int wBlocks   = (D_ * D_ / 8) / 256;   //  512
  cvt_f32_f16<<<actBlocks, 256, 0, stream>>>(query, xq);
  cvt_f32_f16<<<actBlocks, 256, 0, stream>>>(key,   xk);
  cvt_f32_f16<<<actBlocks, 256, 0, stream>>>(value, xv);
  cvt_f32_f16<<<wBlocks, 256, 0, stream>>>(Wq, wq16);
  cvt_f32_f16<<<wBlocks, 256, 0, stream>>>(Wk, wk16);
  cvt_f32_f16<<<wBlocks, 256, 0, stream>>>(Wv, wv16);
  cvt_f32_f16<<<wBlocks, 256, 0, stream>>>(Wo, wo16);

  dim3 ggrid(M_ / 128, D_ / 64);
  gemm_wmma<0><<<ggrid, 256, 0, stream>>>(xq, wq16, bq, qh);
  gemm_wmma<0><<<ggrid, 256, 0, stream>>>(xk, wk16, bk, kh);
  gemm_wmma<0><<<ggrid, 256, 0, stream>>>(xv, wv16, bv, vh);
  attn_wmma<<<dim3(T_ / 64, B_ * H_), 128, 0, stream>>>(qh, kh, vh, ctx);
  gemm_wmma<1><<<ggrid, 256, 0, stream>>>(ctx, wo16, bo, d_out);
}
